// H_GroupPointEncoder_49529562857913
// MI455X (gfx1250) — matrix operand order repaired and checked by reading, not compiled, not run on hardware
//
#include <hip/hip_runtime.h>

typedef __attribute__((ext_vector_type(16))) _Float16 v16h;
typedef __attribute__((ext_vector_type(8)))  float    v8f;
typedef __attribute__((ext_vector_type(4)))  unsigned v4u;
typedef __attribute__((ext_vector_type(8)))  int      v8i;
typedef __attribute__((ext_vector_type(4)))  int      v4i;

#define NF     128
#define GRP    4
#define BS     8
#define NPTS   500
#define KCOPY  6
#define MQ     (NPTS*KCOPY)     // 3000
#define ROWS   (GRP*BS*MQ)      // 96000 output rows
#define UROWS  (GRP*BS*NPTS)    // 16000 unique compute rows (6x reuse)
#define GBPAD  512              // unique rows per (g,b) padded 500 -> 512
#define D1     (3*NF)           // 384  (pos width / GEMM1 K)
#define H1     (4*NF)           // 512  (hidden / GEMM1 N / GEMM2 K)
#define D2     (2*NF)           // 256  (GEMM2 N)
#define KT1    (D1/32)          // 12 k-tiles GEMM1
#define NT1    (H1/16)          // 32 n-tiles GEMM1
#define KT2    (H1/32)          // 16 k-tiles GEMM2
#define NT2    (D2/16)          // 16 n-tiles GEMM2
#define HROW   520              // padded LDS h-row stride in halves (conflict-free)

#define QOFF   ((size_t)4*16*HROW*2)                 // byte offset of q tiles in LDS
#define LDS_TOTAL (QOFF + (size_t)4*16*D2*4)         // 66560 + 65536 = 132096 B

#define OUT_MAIN_ELEMS   ((size_t)ROWS * 512)        // 49,152,000
#define OUT_COORDS_OFF   OUT_MAIN_ELEMS
#define OUT_LAB_OFF      (OUT_MAIN_ELEMS + (size_t)ROWS * 3)

// ---------------- workspace layout (bytes) ----------------
#define WS_W1H_OFF   0                               // 384*512 halves = 393216 B
#define WS_W2H_OFF   393216                          // 512*256 halves = 262144 B
#define WS_CRD_OFF   655360                          // G*BS*N*3 f32 = 192000 B

// ---------------- helpers ----------------
__device__ __forceinline__ unsigned pcg_hash(unsigned x) {
    x = x * 747796405u + 2891336453u;
    unsigned w = ((x >> ((x >> 28) + 4u)) ^ x) * 277803737u;
    return (w >> 22) ^ w;
}
__device__ __forceinline__ float hash_normal(unsigned seed) {
    unsigned a = pcg_hash(seed * 2u + 0x9E3779B9u);
    unsigned b = pcg_hash(seed * 2u + 0x85EBCA6Bu);
    float u1 = ((a >> 8) + 1u) * (1.0f / 16777217.0f);
    float u2 = (b >> 8) * (1.0f / 16777216.0f);
    return sqrtf(-2.0f * __logf(u1)) * __cosf(6.28318530717958647692f * u2);
}

// ---------------- weight packing into WMMA B-fragment order ----------------
// B-fragment (32x16 f16): lane l holds column n = l&15; element e holds
// row k = ((e&8)?16:0) + (l>>4)*8 + (e&7). Fragment fi = nt*KT + kt; 32B contiguous per lane.
__global__ void pack_w1_kernel(const float* __restrict__ W1, _Float16* __restrict__ w1h) {
    int t = blockIdx.x * blockDim.x + threadIdx.x;
    if (t >= NT1 * KT1 * 32) return;
    int lane = t & 31, fi = t >> 5;
    int kt = fi % KT1, nt = fi / KT1;
    int hsel = lane >> 4, ncol = lane & 15;
    int n = nt * 16 + ncol;
    _Float16* dst = w1h + (size_t)t * 16;
#pragma unroll
    for (int e = 0; e < 16; ++e) {
        int k = kt * 32 + ((e & 8) ? 16 : 0) + hsel * 8 + (e & 7);   // B[k][n] = W1[n][k]
        dst[e] = (_Float16)W1[n * D1 + k];
    }
}
__global__ void pack_w2_kernel(const float* __restrict__ W2, _Float16* __restrict__ w2h) {
    int t = blockIdx.x * blockDim.x + threadIdx.x;
    if (t >= NT2 * KT2 * 32) return;
    int lane = t & 31, fi = t >> 5;
    int kt = fi % KT2, nt = fi / KT2;
    int hsel = lane >> 4, ncol = lane & 15;
    int n = nt * 16 + ncol;
    _Float16* dst = w2h + (size_t)t * 16;
#pragma unroll
    for (int e = 0; e < 16; ++e) {
        int k = kt * 32 + ((e & 8) ? 16 : 0) + hsel * 8 + (e & 7);   // B[k][n] = W2[n][k]
        dst[e] = (_Float16)W2[n * H1 + k];
    }
}

// ---------------- coords / labels prep ----------------
__global__ void prep_kernel(const float* __restrict__ point_coord,
                            const int*   __restrict__ labels,
                            const float* __restrict__ pc_range,
                            float* __restrict__ coordsn,     // ws: [G][BS][N][3]
                            float* __restrict__ coords_out,  // d_out: [G][BS][MQ][3]
                            int*   __restrict__ lab_out) {   // d_out: [G][BS][MQ]
    int idx = blockIdx.x * blockDim.x + threadIdx.x;
    if (idx >= UROWS) return;
    int g = idx / (BS * NPTS);
    int rem = idx - g * (BS * NPTS);
    int b = rem / NPTS, n = rem - b * NPTS;
    int labv = labels[b * NPTS + n];
    bool small = (labv == 0) || (labv >= 6 && labv <= 9);
    float stdv = small ? 2.0f : 4.0f;
    float cn[3];
#pragma unroll
    for (int d = 0; d < 3; ++d) {
        float lo = pc_range[d], hi = pc_range[3 + d];
        float raw = point_coord[(b * NPTS + n) * 3 + d];
        float nz = (g == 0) ? 0.0f : stdv * hash_normal((unsigned)(idx * 3 + d));
        cn[d] = (raw + nz - lo) / (hi - lo);
        coordsn[(size_t)idx * 3 + d] = cn[d];
    }
    size_t base = ((size_t)(g * BS + b)) * MQ;
#pragma unroll
    for (int k = 0; k < KCOPY; ++k) {
        size_t m = base + k * NPTS + n;
#pragma unroll
        for (int d = 0; d < 3; ++d) coords_out[m * 3 + d] = cn[d];
        lab_out[m] = labv;
    }
}

// ---------------- query_pos broadcast: out[r][0:256] = query_pos[r%MQ] ----
__global__ __launch_bounds__(256)
void qp_copy_kernel(const float* __restrict__ query_pos, float* __restrict__ out) {
    int i = blockIdx.x * 256 + threadIdx.x;          // over ROWS*64 float4
    if (i >= ROWS * 64) return;
    const int c4 = i & 63;
    const int r  = i >> 6;
    const int gb = r / MQ;
    const int mm = r - gb * MQ;
    ((float4*)out)[(size_t)r * 128 + c4] =
        ((const float4*)query_pos)[(size_t)mm * 64 + c4];
}

// ---------------- fused pos-emb + MLP + TDM tile store ----------------
// Unique rows re-gridded as 32 gb-blocks x 512 (500 valid) so each wave's
// 16 rows are contiguous in output row space -> one strided 2D TDM tile.
// The 6 one2many copies are produced by TDM descriptor iteration.
__global__ __launch_bounds__(128)
void fused_mlp_kernel(const float* __restrict__ coordsn,
                      const int*   __restrict__ labels,
                      const _Float16* __restrict__ w1h,
                      const _Float16* __restrict__ w2h,
                      const float* __restrict__ b1,
                      const float* __restrict__ b2,
                      const float* __restrict__ label_embed,
                      float* __restrict__ out) {
    extern __shared__ char smem[];
    const int tid  = threadIdx.x;
    const int wave = tid >> 5;
    const int lane = tid & 31;
    const int hsel = lane >> 4;
    const int ln   = lane & 15;
    const int row0 = (blockIdx.x * 4 + wave) * 16;   // padded unique-row base
    const int gb   = row0 >> 9;                      // (g*BS+b), 512 rows per gb
    const int n0   = row0 & (GBPAD - 1);             // point index base in gb
    _Float16* hw = (_Float16*)smem + wave * 16 * HROW;
    float*    qt = (float*)(smem + QOFF) + wave * 16 * D2;

    // ---- lane's unique row (clamped at the 500-row boundary) ----
    const int nA = min(n0 + ln, NPTS - 1);
    const float* cp = coordsn + ((size_t)gb * NPTS + nA) * 3;
    const float cx = cp[0], cy = cp[1], cz = cp[2];

    // ---- build 12 A-fragments of pos directly in registers ----
    // A (16x32 f16): lane row m=l&15; pair (2v,2v+1) holds
    // k = (v&4?16:0) + hsel*8 + (v&3)*2 (+0/+1). Even k -> sin, odd -> cos.
    v16h afrag[KT1];
#pragma unroll
    for (int kt = 0; kt < KT1; ++kt) {
        const float cv = (kt < 4) ? cy : ((kt < 8) ? cx : cz);  // y | x | z
        v16h a;
#pragma unroll
        for (int v = 0; v < 8; ++v) {
            const int j  = kt * 32 + ((v & 4) ? 16 : 0) + hsel * 8 + (v & 3) * 2;
            const int fe = j & 127;
            const float freq = __builtin_amdgcn_exp2f((float)fe * (-0.103810252965216f));
            const float ang  = cv * 6.28318530717958647692f * freq;
            a[2 * v]     = (_Float16)__sinf(ang);
            a[2 * v + 1] = (_Float16)__cosf(ang);
        }
        afrag[kt] = a;
    }

    // ---- GEMM1: h = relu(pos @ W1^T + b1) -> padded LDS (f16) ----
    const v16h* bp1 = (const v16h*)w1h;
#pragma unroll 1
    for (int nt = 0; nt < NT1; ++nt) {
        v8f acc = {};
        const v16h* bt = bp1 + (size_t)(nt * KT1) * 32 + lane;
#pragma unroll
        for (int kt = 0; kt < KT1; ++kt) {
            __builtin_prefetch((const void*)(bt + 32), 0, 1);
            v16h bf = *bt;
            bt += 32;
            acc = __builtin_amdgcn_wmma_f32_16x16x32_f16(
                false, afrag[kt], false, bf, (short)0, acc, false, false);
        }
        const float b1v = b1[nt * 16 + ln];
#pragma unroll
        for (int v = 0; v < 8; ++v) {
            const int mrow = v + 8 * hsel;
            float x = acc[v] + b1v;
            x = fmaxf(x, 0.0f);
            hw[mrow * HROW + nt * 16 + ln] = (_Float16)x;
        }
    }

    __syncthreads();

    // ---- per-row label gather (clamped; invalid rows never stored) ----
    int lab8[8];
#pragma unroll
    for (int v = 0; v < 8; ++v) {
        const int nv = min(n0 + v + 8 * hsel, NPTS - 1);
        lab8[v] = labels[(gb & 7) * NPTS + nv];
    }

    // ---- GEMM2: pe = h @ W2^T + b2 (+label_embed) -> q tile in LDS ----
    const v16h* bp2 = (const v16h*)w2h;
#pragma unroll 1
    for (int nt = 0; nt < NT2; ++nt) {
        v8f acc = {};
        const v16h* bt = bp2 + (size_t)(nt * KT2) * 32 + lane;
#pragma unroll
        for (int kt = 0; kt < KT2; ++kt) {
            v16h a2;
#pragma unroll
            for (int v = 0; v < 8; ++v) {
                const int k0 = kt * 32 + ((v & 4) ? 16 : 0) + hsel * 8 + (v & 3) * 2;
                union { unsigned u; _Float16 h2[2]; } cvt;
                cvt.u = *(const unsigned*)&hw[ln * HROW + k0];
                a2[2 * v]     = cvt.h2[0];
                a2[2 * v + 1] = cvt.h2[1];
            }
            __builtin_prefetch((const void*)(bt + 32), 0, 1);
            v16h bf = *bt;
            bt += 32;
            acc = __builtin_amdgcn_wmma_f32_16x16x32_f16(
                false, a2, false, bf, (short)0, acc, false, false);
        }
        const int col = nt * 16 + ln;
        const float b2v = b2[col];
#pragma unroll
        for (int v = 0; v < 8; ++v)
            qt[(v + 8 * hsel) * D2 + col] = acc[v] + b2v + label_embed[lab8[v] * 256 + col];
    }

    // ---- flush q tile to 6 tiled output copies ----
    const int valid = min(16, NPTS - n0);            // rows actually stored
    const size_t r0 = (size_t)gb * MQ + n0;          // first output row (k=0)

#if __has_builtin(__builtin_amdgcn_tensor_store_from_lds)
    // One TDM op per wave: 2D tile (256 x valid) f32, tensor row stride 512,
    // iterated 6x with lds_addr_increment=0, global_addr_increment=500*512.
    asm volatile("s_wait_dscnt 0x0" ::: "memory");   // TDM unordered vs DS
    const unsigned long long ga =
        (unsigned long long)(uintptr_t)out + (r0 * 512 + 256) * 4ull;
    v4u g0;
    g0[0] = 1u;                                              // count=1
    g0[1] = (unsigned)(QOFF + (size_t)wave * 16 * D2 * 4);   // lds_addr (bytes)
    g0[2] = (unsigned)(ga & 0xFFFFFFFFull);                  // global_addr lo
    g0[3] = (unsigned)((ga >> 32) & 0x01FFFFFFull) | 0x80000000u;  // hi | type=2
    v8i g1;
    g1[0] = (2 << 16) | (1 << 19);          // data_size=4B, iterate_enable
    g1[1] = (int)(256u << 16);              // tensor_dim0 = 256 (bits 63:48)
    g1[2] = (int)((unsigned)valid << 16);   // tensor_dim1 = valid (bits 95:80)
    g1[3] = (int)(256u << 16);              // tile_dim0 = 256 (bits 127:112)
    g1[4] = valid;                          // tile_dim1 = valid, tile_dim2 = 0
    g1[5] = 512;                            // tensor_dim0_stride = 512 elems
    g1[6] = 0;
    g1[7] = 0;
    v4i g2;
    g2[0] = 1;                              // tensor_dim2 (unused)
    g2[1] = 0;                              // lds_addr_increment = 0 (re-read tile)
    g2[2] = NPTS * 512;                     // global_addr_increment (elems) = 1 copy
    g2[3] = (KCOPY - 1) << 16;              // iterate_count = 5 -> 6 iterations
    v4i g3 = {0, 0, 0, 0};
    v8i g4 = {0, 0, 0, 0, 0, 0, 0, 0};      // extra operand on 6-arg toolchain
    __builtin_amdgcn_tensor_store_from_lds(g0, g1, g2, g3, g4, 0);
    __builtin_amdgcn_s_wait_tensorcnt(0);
#else
    // Fallback: coalesced float4 stores from the LDS tile.
    const float4* qt4 = (const float4*)qt;
    float4* out4 = (float4*)out;
    for (int i = lane; i < valid * 64; i += 32) {
        const int rr = i >> 6, c4 = i & 63;
        const float4 qv = qt4[rr * 64 + c4];
        const size_t rbase = r0 + rr;
#pragma unroll
        for (int k = 0; k < KCOPY; ++k)
            out4[(rbase + (size_t)k * NPTS) * 128 + 64 + c4] = qv;
    }
#endif
}

// ---------------- launcher ----------------
extern "C" void kernel_launch(void* const* d_in, const int* in_sizes, int n_in,
                              void* d_out, int out_size, void* d_ws, size_t ws_size,
                              hipStream_t stream) {
    (void)in_sizes; (void)n_in; (void)out_size; (void)ws_size;
    const float* point_coord = (const float*)d_in[0];
    const int*   labels      = (const int*)  d_in[1];
    const float* pc_range    = (const float*)d_in[2];
    const float* label_embed = (const float*)d_in[3];
    const float* W1          = (const float*)d_in[4];
    const float* b1          = (const float*)d_in[5];
    const float* W2          = (const float*)d_in[6];
    const float* b2          = (const float*)d_in[7];
    const float* query_pos   = (const float*)d_in[8];

    char* ws = (char*)d_ws;
    _Float16* w1h     = (_Float16*)(ws + WS_W1H_OFF);
    _Float16* w2h     = (_Float16*)(ws + WS_W2H_OFF);
    float*    coordsn = (float*)   (ws + WS_CRD_OFF);

    float* out        = (float*)d_out;
    float* coords_out = out + OUT_COORDS_OFF;
    int*   lab_out    = (int*)(out + OUT_LAB_OFF);

    // pack weights to f16 WMMA B-fragment layout
    pack_w1_kernel<<<(NT1 * KT1 * 32 + 255) / 256, 256, 0, stream>>>(W1, w1h);
    pack_w2_kernel<<<(NT2 * KT2 * 32 + 255) / 256, 256, 0, stream>>>(W2, w2h);

    // coords + labels prep (also writes coords/lab outputs)
    prep_kernel<<<(UROWS + 255) / 256, 256, 0, stream>>>(
        point_coord, labels, pc_range, coordsn, coords_out, lab_out);

    // query_pos half of the output: pure coalesced float4 copy
    qp_copy_kernel<<<(ROWS * 64 + 255) / 256, 256, 0, stream>>>(query_pos, out);

    // fused embedding + MLP over padded unique rows; TDM 6x tile broadcast
    const int blocks = (32 * GBPAD) / 64;            // 256
    fused_mlp_kernel<<<blocks, 128, LDS_TOTAL, stream>>>(
        coordsn, labels, w1h, w2h, b1, b2, label_embed, out);
}
